// GraphEmbedding_80049600463368
// MI455X (gfx1250) — matrix operand order repaired and checked by reading, not compiled, not run on hardware
//
#include <hip/hip_runtime.h>

// ---------------------------------------------------------------------------
// TGN graph-embedding for MI455X (gfx1250, wave32, WMMA).
//
// Roofline: gathers ~1.03 GB -> ~45us @ 23.3 TB/s. Projection GEMMs (~271
// GFLOP) run on v_wmma_f32_16x16x32_bf16; only hd=128 score dots and K=20
// context sums (0.66 G MAC total) remain on VALU, vectorized via packed
// bf16 LDS loads. K|V GEMM blocks 2 M-tiles per B-fragment to halve L2
// weight traffic (~16 GB -> ~8 GB). 8 rows per workgroup, ~311 KB LDS
// (<=320 KB/WG per CDNA5 ISA), 256 threads = 8 wave32.
// ---------------------------------------------------------------------------

typedef __bf16 bf16;
typedef __attribute__((ext_vector_type(16))) __bf16 v16bf;
typedef __attribute__((ext_vector_type(8)))  __bf16 v8bf;
typedef __attribute__((ext_vector_type(4)))  __bf16 v4bf;
typedef __attribute__((ext_vector_type(8)))  float  v8f;
typedef __attribute__((ext_vector_type(4)))  float  v4f;

#define DFE   128     // feature dim
#define KNB   20      // neighbors per row
#define TB    8       // batch rows per workgroup
#define PAIRS 160     // TB*KNB
#define KVS   392     // keyv LDS stride (384 + pad, multiple of 8)
#define KPS   520     // k|v projected LDS stride (512 + pad)

// ---- workspace layout (bf16 element offsets; all multiples of 64) ----
#define WS_WQ1 0                    // [256][128]  Wq[:, :128]
#define WS_WK  32768                // [256][384]
#define WS_WV  131072               // [256][384]
#define WS_WO  229376               // [256][256]
#define WS_W1  294912               // [128][384]
#define WS_W2  344064               // [128][128]
#define WS_BF_TOTAL 360448
#define WS_QBIAS_BYTES (WS_BF_TOTAL * 2)   // f32[256]: Wq[:,128:]@cos(b_t)+bq

// ---- dynamic LDS layout (bytes) ----
#define L_KV    0            // bf16 [160][392] = 125440 (reused after phase 3)
#define L_KVP   125440       // bf16 [160][520] = 166400
#define L_SRC   291840       // f32  [8][128]
#define L_SRCQ  295936       // bf16 [16][136]  (q GEMM A-matrix, rows 8..15 zero)
#define L_WT    300288       // f32[128]
#define L_BT    300800       // f32[128]
#define L_Q     301312       // f32  [8][256]
#define L_NBR   309504       // int[160]
#define L_EIDX  310144       // int[160]
#define L_ET    310784       // f32[160]
#define L_TS    311424       // f32[8]
#define L_INV   311456       // int[8]
#define L_TOTAL 311488
// overlay inside [0,125440) -- valid after K/V GEMM:
#define L_SCORES 0           // f32 [8][2][20]
#define L_CTXA   2048        // bf16 [16][264]
#define L_FC1A   10496       // bf16 [16][392]
#define L_FC2A   23040       // bf16 [16][136]
#define L_OVL_END 27392

__device__ __forceinline__ bf16 f2bf(float f) {
  unsigned u = __builtin_bit_cast(unsigned, f);
  unsigned r = u + 0x7fffu + ((u >> 16) & 1u);            // round-nearest-even
  unsigned short h = (unsigned short)(r >> 16);
  return __builtin_bit_cast(bf16, h);
}
__device__ __forceinline__ float bf2f(bf16 b) {
  unsigned u = ((unsigned)__builtin_bit_cast(unsigned short, b)) << 16;
  return __builtin_bit_cast(float, u);
}

// A-fragment (16x32 bf16). Lane l: M=l&15, halves j -> K=kb+(l>>4)*16+j
// (contiguous, matching ldB).
__device__ __forceinline__ v16bf ldsA(const bf16* base, int ld, int kb, int lane) {
  const bf16* p = base + (lane & 15) * ld + kb + ((lane >> 4) << 4);
  v8bf lo = *(const v8bf*)p;
  v8bf hi = *(const v8bf*)(p + 8);
  return __builtin_shufflevector(lo, hi, 0,1,2,3,4,5,6,7,8,9,10,11,12,13,14,15);
}
// B-fragment (32x16 bf16) from row-major-by-N weight W[n][k]:
// lane l: N = n0 + (l&15), halves j -> K = kb + (l>>4)*16 + j (contiguous 32B).
__device__ __forceinline__ v16bf ldB(const bf16* W, int ld, int n0, int kb, int lane) {
  return *(const v16bf*)(W + (size_t)(n0 + (lane & 15)) * ld + kb + ((lane >> 4) << 4));
}
__device__ __forceinline__ v8f wmma_b(v16bf a, v16bf b, v8f c) {
  return __builtin_amdgcn_wmma_f32_16x16x32_bf16(false, a, false, b, (short)0, c,
                                                 false, false);
}

// ---------------------------------------------------------------------------
// Prep: fp32 weights -> bf16 in d_ws, plus folded q-bias (constant time enc).
// ---------------------------------------------------------------------------
__global__ void tgn_prep(const float* __restrict__ Wq, const float* __restrict__ bq,
                         const float* __restrict__ Wk, const float* __restrict__ Wv,
                         const float* __restrict__ Wo, const float* __restrict__ W1,
                         const float* __restrict__ W2, const float* __restrict__ b_t,
                         bf16* __restrict__ wsb, float* __restrict__ qbias) {
  int gid = blockIdx.x * blockDim.x + threadIdx.x;
  int stride = gridDim.x * blockDim.x;
  for (int i = gid; i < WS_BF_TOTAL; i += stride) {
    float v;
    if (i < WS_WK)       v = Wq[(i >> 7) * 256 + (i & 127)];   // Wq[:, :128]
    else if (i < WS_WV)  v = Wk[i - WS_WK];
    else if (i < WS_WO)  v = Wv[i - WS_WV];
    else if (i < WS_W1)  v = Wo[i - WS_WO];
    else if (i < WS_W2)  v = W1[i - WS_W1];
    else                 v = W2[i - WS_W2];
    wsb[i] = f2bf(v);
  }
  if (gid < 256) {     // qbias[n] = bq[n] + sum_j Wq[n][128+j]*cos(b_t[j])
    float s = bq[gid];
    for (int j = 0; j < 128; ++j) s += Wq[gid * 256 + 128 + j] * cosf(b_t[j]);
    qbias[gid] = s;
  }
}

// ---------------------------------------------------------------------------
// Main fused kernel: one workgroup (256 thr = 8 wave32) per 8 batch rows.
// ---------------------------------------------------------------------------
__global__ __launch_bounds__(256, 1)
void tgn_main(const float* __restrict__ nodef, const float* __restrict__ edgef,
              const float* __restrict__ memf,  const float* __restrict__ w_t,
              const float* __restrict__ b_t,   const float* __restrict__ bk,
              const float* __restrict__ bv,    const float* __restrict__ bo,
              const float* __restrict__ b1,    const float* __restrict__ b2,
              const float* __restrict__ tsg,   const float* __restrict__ etg,
              const int* __restrict__ srcn,    const int* __restrict__ neigh,
              const int* __restrict__ eidxg,   const bf16* __restrict__ wsb,
              const float* __restrict__ qbias, float* __restrict__ outp, int Btot) {
  extern __shared__ char smem[];
  bf16*  kv   = (bf16*)(smem + L_KV);
  bf16*  kvp  = (bf16*)(smem + L_KVP);
  float* srcf = (float*)(smem + L_SRC);
  bf16*  srcq = (bf16*)(smem + L_SRCQ);
  float* wt   = (float*)(smem + L_WT);
  float* bt   = (float*)(smem + L_BT);
  float* qf   = (float*)(smem + L_Q);
  int*   nbrL = (int*)(smem + L_NBR);
  int*   eixL = (int*)(smem + L_EIDX);
  float* etL  = (float*)(smem + L_ET);
  float* tsL  = (float*)(smem + L_TS);
  int*   invL = (int*)(smem + L_INV);
  float* scores = (float*)(smem + L_SCORES);
  bf16*  ctxA = (bf16*)(smem + L_CTXA);
  bf16*  fc1A = (bf16*)(smem + L_FC1A);
  bf16*  fc2A = (bf16*)(smem + L_FC2A);

  const bf16* wq1 = wsb + WS_WQ1;
  const bf16* wkb = wsb + WS_WK;
  const bf16* wvb = wsb + WS_WV;
  const bf16* wob = wsb + WS_WO;
  const bf16* w1b = wsb + WS_W1;
  const bf16* w2b = wsb + WS_W2;

  const int tid  = threadIdx.x;
  const int wave = tid >> 5;
  const int lane = tid & 31;
  const int b0   = blockIdx.x * TB;

  // ---- Phase 0: metadata, time-enc params, zero q-GEMM A pad --------------
  for (int p = tid; p < PAIRS; p += 256) {
    int row = p / KNB, k = p % KNB;
    int br = b0 + row; if (br >= Btot) br = Btot - 1;
    size_t gi = (size_t)br * KNB + k;
    nbrL[p] = neigh[gi];
    eixL[p] = eidxg[gi];
    etL[p]  = etg[gi];
  }
  if (tid < TB) { int br = b0 + tid; if (br >= Btot) br = Btot - 1; tsL[tid] = tsg[br]; }
  if (tid < 128) { wt[tid] = w_t[tid]; bt[tid] = b_t[tid]; }
  for (int i = tid; i < (16 * 136 * 2) / 4; i += 256) ((int*)(smem + L_SRCQ))[i] = 0;
  __syncthreads();

  // ---- Phase 1: gather + build keyv (bf16) and src features ---------------
  if (tid < TB) {                                   // invalid-row flags
    int all0 = 1;
    for (int k = 0; k < KNB; ++k) all0 &= (nbrL[tid * KNB + k] == 0);
    invL[tid] = all0;
  }
  for (int p = wave; p < PAIRS + TB; p += 8) {      // uniform per wave
    if (p < PAIRS) {
      int nbr = nbrL[p], eix = eixL[p];
      int row = p / KNB;
      float delta = tsL[row] - etL[p];
      int pn = p + 8;                               // prefetch next gathers
      if (pn < PAIRS) {
        __builtin_prefetch(nodef + (size_t)nbrL[pn] * DFE + lane * 4, 0, 0);
        __builtin_prefetch(memf  + (size_t)nbrL[pn] * DFE + lane * 4, 0, 0);
        __builtin_prefetch(edgef + (size_t)eixL[pn] * DFE + lane * 4, 0, 0);
      }
      v4f nf = *(const v4f*)(nodef + (size_t)nbr * DFE + lane * 4);
      v4f mm = *(const v4f*)(memf  + (size_t)nbr * DFE + lane * 4);
      v4f ef = *(const v4f*)(edgef + (size_t)eix * DFE + lane * 4);
      bf16* kp = kv + p * KVS;
      v4bf kn, kt, ke;
#pragma unroll
      for (int i = 0; i < 4; ++i) {
        int j = lane * 4 + i;
        kn[i] = f2bf(nf[i] + mm[i]);
        kt[i] = f2bf(cosf(delta * wt[j] + bt[j]));
        ke[i] = f2bf(ef[i]);
      }
      *(v4bf*)(kp + lane * 4)       = kn;           // packed 8B LDS stores
      *(v4bf*)(kp + 128 + lane * 4) = kt;
      *(v4bf*)(kp + 256 + lane * 4) = ke;
    } else {
      int row = p - PAIRS;
      int br = b0 + row; if (br >= Btot) br = Btot - 1;
      int sn = srcn[br];
      v4f nf = *(const v4f*)(nodef + (size_t)sn * DFE + lane * 4);
      v4f mm = *(const v4f*)(memf  + (size_t)sn * DFE + lane * 4);
      v4f sv; v4bf sb;
#pragma unroll
      for (int i = 0; i < 4; ++i) {
        float s = nf[i] + mm[i];
        sv[i] = s; sb[i] = f2bf(s);
      }
      *(v4f*)(srcf + row * 128 + lane * 4)  = sv;
      *(v4bf*)(srcq + row * 136 + lane * 4) = sb;
    }
  }
  __syncthreads();

  // ---- Phase 2 (WMMA): q = src_feat @ Wq[:, :128].T + folded bias ---------
  {
    v8f a0 = {}, a1 = {};
    int n0 = wave * 32, n1 = n0 + 16;
    for (int ks = 0; ks < 4; ++ks) {
      v16bf a = ldsA(srcq, 136, ks * 32, lane);
      a0 = wmma_b(a, ldB(wq1, 128, n0, ks * 32, lane), a0);
      a1 = wmma_b(a, ldB(wq1, 128, n1, ks * 32, lane), a1);
    }
    int c0 = n0 + (lane & 15), c1 = n1 + (lane & 15);
    float qb0 = qbias[c0], qb1 = qbias[c1];
#pragma unroll
    for (int r = 0; r < 8; ++r) {
      int m = r + ((lane >> 4) << 3);
      if (m < TB) { qf[m * 256 + c0] = a0[r] + qb0; qf[m * 256 + c1] = a1[r] + qb1; }
    }
  }
  __syncthreads();

  // ---- Phase 3 (WMMA): k|v = keyv @ [Wk|Wv].T + bias, 160x384x512 ---------
  // 2 M-tiles per B-fragment: halves weight (L2) traffic vs per-M reload.
  {
    const bf16* Wsel[4]; int nsel[4]; float bias[4];
#pragma unroll
    for (int i = 0; i < 4; ++i) {
      int col0 = (wave * 4 + i) * 16;
      int col = col0 + (lane & 15);
      if (col0 < 256) { Wsel[i] = wkb; nsel[i] = col0;       bias[i] = bk[col]; }
      else            { Wsel[i] = wvb; nsel[i] = col0 - 256; bias[i] = bv[col - 256]; }
    }
    for (int mtb = 0; mtb < 5; ++mtb) {
      v8f acc0[4] = {}, acc1[4] = {};
      for (int ks = 0; ks < 12; ++ks) {
        v16bf a0 = ldsA(kv + (mtb * 2 + 0) * 16 * KVS, KVS, ks * 32, lane);
        v16bf a1 = ldsA(kv + (mtb * 2 + 1) * 16 * KVS, KVS, ks * 32, lane);
#pragma unroll
        for (int i = 0; i < 4; ++i) {
          v16bf b = ldB(Wsel[i], 384, nsel[i], ks * 32, lane);
          acc0[i] = wmma_b(a0, b, acc0[i]);
          acc1[i] = wmma_b(a1, b, acc1[i]);
        }
      }
#pragma unroll
      for (int i = 0; i < 4; ++i) {
        int col = (wave * 4 + i) * 16 + (lane & 15);
#pragma unroll
        for (int r = 0; r < 8; ++r) {
          int m = r + ((lane >> 4) << 3);
          int p0 = (mtb * 2 + 0) * 16 + m, p1 = (mtb * 2 + 1) * 16 + m;
          kvp[p0 * KPS + col] = f2bf(acc0[i][r] + bias[i]);
          kvp[p1 * KPS + col] = f2bf(acc1[i][r] + bias[i]);
        }
      }
    }
  }
  __syncthreads();

  // ---- Phase 3b: zero overlay region (keyv no longer needed) --------------
  for (int i = tid; i < L_OVL_END / 4; i += 256) ((int*)smem)[i] = 0;
  __syncthreads();

  // ---- Phase 4: attention scores (hd=128 dots, packed loads) + src->fc1A --
  for (int d = tid; d < TB * 2 * KNB; d += 256) {
    int row = d / 40, rem = d % 40, h = rem / 20, k = rem % 20;
    int pair = row * KNB + k;
    const v8bf* kp8 = (const v8bf*)(kvp + pair * KPS + h * 128);   // 16B aligned
    const v4f*  qp4 = (const v4f*)(qf + row * 256 + h * 128);      // 16B aligned
    float s = 0.f;
#pragma unroll 4
    for (int jj = 0; jj < 16; ++jj) {
      v8bf kb = kp8[jj];
      v4f q0 = qp4[jj * 2], q1 = qp4[jj * 2 + 1];
      s += q0[0] * bf2f(kb[0]) + q0[1] * bf2f(kb[1]) +
           q0[2] * bf2f(kb[2]) + q0[3] * bf2f(kb[3]) +
           q1[0] * bf2f(kb[4]) + q1[1] * bf2f(kb[5]) +
           q1[2] * bf2f(kb[6]) + q1[3] * bf2f(kb[7]);
    }
    s *= 0.08838834764831845f;                       // 1/sqrt(128)
    bool msk = (nbrL[pair] == 0) && !(k == 0 && invL[row]);
    scores[row * 40 + h * 20 + k] = msk ? -1e9f : s;
  }
  for (int o = tid; o < TB * 128; o += 256) {        // fc1A[:, 256:384] = src
    int row = o >> 7, j = o & 127;
    fc1A[row * 392 + 256 + j] = f2bf(srcf[row * 128 + j]);
  }
  __syncthreads();

  // ---- Phase 5: softmax over K per (row, head) ----------------------------
  if (tid < TB * 2) {
    float* s = scores + tid * 20;
    float mx = s[0];
    for (int k = 1; k < KNB; ++k) mx = fmaxf(mx, s[k]);
    float sum = 0.f;
    for (int k = 0; k < KNB; ++k) { float e = expf(s[k] - mx); s[k] = e; sum += e; }
    float inv = 1.f / sum;
    for (int k = 0; k < KNB; ++k) s[k] *= inv;
  }
  __syncthreads();

  // ---- Phase 6: ctx = attn @ v (K=20 weighted sum, packed loads) ----------
  for (int o4 = tid; o4 < TB * 64; o4 += 256) {      // 4 columns per thread
    int row = o4 >> 6, c4 = (o4 & 63) * 4, h = c4 >> 7;
    const float* a = scores + row * 40 + h * 20;
    float s0 = 0.f, s1 = 0.f, s2 = 0.f, s3 = 0.f;
    for (int k = 0; k < KNB; ++k) {
      v4bf vv = *(const v4bf*)(kvp + (row * KNB + k) * KPS + 256 + c4);  // 8B aligned
      float w = a[k];
      s0 += w * bf2f(vv[0]); s1 += w * bf2f(vv[1]);
      s2 += w * bf2f(vv[2]); s3 += w * bf2f(vv[3]);
    }
    v4bf cv; cv[0] = f2bf(s0); cv[1] = f2bf(s1); cv[2] = f2bf(s2); cv[3] = f2bf(s3);
    *(v4bf*)(ctxA + row * 264 + c4) = cv;
  }
  __syncthreads();

  // ---- Phase 7 (WMMA): out = ctx @ Wo.T + bo (zero invalid rows) ----------
  {
    v8f a0 = {}, a1 = {};
    int n0 = wave * 32, n1 = n0 + 16;
    for (int ks = 0; ks < 8; ++ks) {
      v16bf a = ldsA(ctxA, 264, ks * 32, lane);
      a0 = wmma_b(a, ldB(wob, 256, n0, ks * 32, lane), a0);
      a1 = wmma_b(a, ldB(wob, 256, n1, ks * 32, lane), a1);
    }
    int c0 = n0 + (lane & 15), c1 = n1 + (lane & 15);
    float bo0 = bo[c0], bo1 = bo[c1];
#pragma unroll
    for (int r = 0; r < 8; ++r) {
      int m = r + ((lane >> 4) << 3);
      if (m < TB) {
        float o0 = a0[r] + bo0, o1 = a1[r] + bo1;
        if (invL[m]) { o0 = 0.f; o1 = 0.f; }
        fc1A[m * 392 + c0] = f2bf(o0);
        fc1A[m * 392 + c1] = f2bf(o1);
      }
    }
  }
  __syncthreads();

  // ---- Phase 8 (WMMA): h = relu([out|src] @ W1.T + b1) --------------------
  {
    v8f acc = {};
    int n0 = wave * 16;
    for (int ks = 0; ks < 12; ++ks) {
      v16bf a = ldsA(fc1A, 392, ks * 32, lane);
      acc = wmma_b(a, ldB(w1b, 384, n0, ks * 32, lane), acc);
    }
    int col = n0 + (lane & 15);
    float b = b1[col];
#pragma unroll
    for (int r = 0; r < 8; ++r) {
      int m = r + ((lane >> 4) << 3);
      if (m < TB) { float o = acc[r] + b; fc2A[m * 136 + col] = f2bf(o > 0.f ? o : 0.f); }
    }
  }
  __syncthreads();

  // ---- Phase 9 (WMMA): final = h @ W2.T + b2 -> global --------------------
  {
    v8f acc = {};
    int n0 = wave * 16;
    for (int ks = 0; ks < 4; ++ks) {
      v16bf a = ldsA(fc2A, 136, ks * 32, lane);
      acc = wmma_b(a, ldB(w2b, 128, n0, ks * 32, lane), acc);
    }
    int col = n0 + (lane & 15);
    float b = b2[col];
#pragma unroll
    for (int r = 0; r < 8; ++r) {
      int m = r + ((lane >> 4) << 3);
      if (m < TB && b0 + m < Btot) outp[(size_t)(b0 + m) * 128 + col] = acc[r] + b;
    }
  }
}

extern "C" void kernel_launch(void* const* d_in, const int* in_sizes, int n_in,
                              void* d_out, int out_size, void* d_ws, size_t ws_size,
                              hipStream_t stream) {
  (void)n_in; (void)out_size; (void)ws_size;
  const float* nodef = (const float*)d_in[0];
  const float* edgef = (const float*)d_in[1];
  const float* memf  = (const float*)d_in[2];
  const float* w_t   = (const float*)d_in[3];
  const float* b_t   = (const float*)d_in[4];
  const float* Wq = (const float*)d_in[5];
  const float* bq = (const float*)d_in[6];
  const float* Wk = (const float*)d_in[7];
  const float* bk = (const float*)d_in[8];
  const float* Wv = (const float*)d_in[9];
  const float* bv = (const float*)d_in[10];
  const float* Wo = (const float*)d_in[11];
  const float* bo = (const float*)d_in[12];
  const float* W1 = (const float*)d_in[13];
  const float* b1 = (const float*)d_in[14];
  const float* W2 = (const float*)d_in[15];
  const float* b2 = (const float*)d_in[16];
  const float* ts = (const float*)d_in[17];
  const float* et = (const float*)d_in[18];
  const int* srcn  = (const int*)d_in[19];
  const int* neigh = (const int*)d_in[20];
  const int* eidx  = (const int*)d_in[21];
  int Btot = in_sizes[17];

  bf16*  wsb   = (bf16*)d_ws;
  float* qbias = (float*)((char*)d_ws + WS_QBIAS_BYTES);
  float* outp  = (float*)d_out;

  tgn_prep<<<(WS_BF_TOTAL + 255) / 256, 256, 0, stream>>>(Wq, bq, Wk, Wv, Wo, W1, W2,
                                                          b_t, wsb, qbias);

  (void)hipFuncSetAttribute(reinterpret_cast<const void*>(tgn_main),
                            hipFuncAttributeMaxDynamicSharedMemorySize, L_TOTAL);
  int grid = (Btot + TB - 1) / TB;
  tgn_main<<<grid, 256, L_TOTAL, stream>>>(nodef, edgef, memf, w_t, b_t, bk, bv, bo,
                                           b1, b2, ts, et, srcn, neigh, eidx,
                                           wsb, qbias, outp, Btot);
}
